// GatedGraphConvolution_37907381354546
// MI455X (gfx1250) — matrix operand order repaired and checked by reading, not compiled, not run on hardware
//
#include <hip/hip_runtime.h>
#include <math.h>

// ---------------------------------------------------------------------------
// GatedGraphConvolution on MI455X (gfx1250): bf16 WMMA (f32 accum) pipeline.
// All GEMM operands are pre-converted to bf16 so the hot loops are pure
// global_load_b128 + v_wmma_f32_16x16x32_bf16 (no f32->bf16 cvt in loop).
// ---------------------------------------------------------------------------

typedef __attribute__((ext_vector_type(16))) __bf16 bf16x16;
typedef __attribute__((ext_vector_type(8)))  __bf16 bf16x8;
typedef __attribute__((ext_vector_type(8)))  float  f32x8;
typedef __attribute__((ext_vector_type(4)))  float  f32x4;

#define WMMA_BF16(a, b, c) \
  __builtin_amdgcn_wmma_f32_16x16x32_bf16(false, (a), false, (b), (short)0, (c), false, false)

// ---- fragment loaders -----------------------------------------------------
// A fragment (16x32 bf16, row-major source [.., ld]):
//   lane L: row = row0 + L%16 ; elems 0..7  -> K = kb + (L/16)*8 + j
//                               elems 8..15 -> K = kb + 16 + (L/16)*8 + j
static __device__ __forceinline__ bf16x16
load_a_bf16(const __bf16* base, int row0, int kb, int ld, int lane) {
  int r  = row0 + (lane & 15);
  int kg = (lane >> 4) * 8;
  const __bf16* p = base + (size_t)r * ld + (kb + kg);
  bf16x8 lo = *(const bf16x8*)(p);        // 16B
  bf16x8 hi = *(const bf16x8*)(p + 16);   // 16B
  bf16x16 out;
#pragma unroll
  for (int j = 0; j < 8; ++j) { out[j] = lo[j]; out[j + 8] = hi[j]; }
  return out;
}

// B fragment (32x16 bf16) from a pre-transposed B^T [N, K] row-major:
//   lane L: col n = col0 + L%16 ; elems 0..7 -> K = kb + (L/16)*8 + j, elems 8..15 -> K = kb + 16 + (L/16)*8 + j
static __device__ __forceinline__ bf16x16
load_b_T(const __bf16* bt, int col0, int kb, int ldk, int lane) {
  int n = col0 + (lane & 15);
  int k = kb + (lane >> 4) * 8;
  const __bf16* p = bt + (size_t)n * ldk + k;
  bf16x8 lo = *(const bf16x8*)(p);
  bf16x8 hi = *(const bf16x8*)(p + 16);
  bf16x16 out;
#pragma unroll
  for (int j = 0; j < 8; ++j) { out[j] = lo[j]; out[j + 8] = hi[j]; }
  return out;
}

static __device__ __forceinline__ float sigmoidf_(float x) {
  return 1.0f / (1.0f + __expf(-x));
}

// ---------------------------------------------------------------------------
// Elementwise f32 -> bf16 convert (row-major preserved). n multiple of 8.
// ---------------------------------------------------------------------------
__global__ void __launch_bounds__(256)
ggc_cvt_bf16(const float* __restrict__ src, __bf16* __restrict__ dst, size_t n) {
  size_t i = ((size_t)blockIdx.x * blockDim.x + threadIdx.x) * 8;
  if (i >= n) return;
  f32x4 a = *(const f32x4*)(src + i);
  f32x4 b = *(const f32x4*)(src + i + 4);
  bf16x8 o;
#pragma unroll
  for (int j = 0; j < 4; ++j) { o[j] = (__bf16)a[j]; o[j + 4] = (__bf16)b[j]; }
  *(bf16x8*)(dst + i) = o;
}

// ---------------------------------------------------------------------------
// LDS-tiled transpose + f32->bf16 convert:  dst[C,R] = (bf16) src[R,C]
// gridDim.z = batch (stride R*C both sides). block (32,8).
// ---------------------------------------------------------------------------
__global__ void ggc_transpose_bf16(const float* __restrict__ src,
                                   __bf16* __restrict__ dst, int R, int C) {
  __shared__ __bf16 tile[32][33];
  const float* s = src + (size_t)blockIdx.z * R * C;
  __bf16*      d = dst + (size_t)blockIdx.z * R * C;
  int x  = blockIdx.x * 32 + threadIdx.x;       // src col
  int y0 = blockIdx.y * 32;                     // src row base
#pragma unroll
  for (int i = 0; i < 32; i += 8)
    tile[threadIdx.y + i][threadIdx.x] = (__bf16)s[(size_t)(y0 + threadIdx.y + i) * C + x];
  __syncthreads();
  int x2 = blockIdx.y * 32 + threadIdx.x;       // dst col (= src row)
  int y2 = blockIdx.x * 32;                     // dst row base (= src col)
#pragma unroll
  for (int i = 0; i < 32; i += 8)
    d[(size_t)(y2 + threadIdx.y + i) * R + x2] = tile[threadIdx.x][threadIdx.y + i];
}

// ---------------------------------------------------------------------------
// Kernel A: msg[b] = Adj[b](256x256) @ H[b](256x512 via H^T) + gc_bias
// 32x16 tile per wave (2 WMMAs/K-step, B-frag reused for 2 row tiles).
// 128 * 8 * 32 = 32768 waves -> 4096 blocks.
// ---------------------------------------------------------------------------
__global__ void __launch_bounds__(256)
ggc_gconv(const __bf16* __restrict__ adj_bf, const __bf16* __restrict__ hT,
          const float* __restrict__ gc_bias, __bf16* __restrict__ msg_bf) {
  const int lane = threadIdx.x & 31;
  const int t    = blockIdx.x * 8 + (threadIdx.x >> 5);
  const int b    = t >> 8;                 // 256 tiles per batch
  const int rem  = t & 255;
  const int r0   = (rem >> 5) * 32;        // 8 row groups of 32
  const int col0 = (rem & 31) * 16;        // 32 col tiles

  const __bf16* A  = adj_bf + (size_t)b * 256 * 256;
  const __bf16* BT = hT     + (size_t)b * 512 * 256;

  f32x8 acc0 = {}, acc1 = {};
#pragma unroll 4
  for (int kb = 0; kb < 256; kb += 32) {
    bf16x16 bf = load_b_T(BT, col0, kb, 256, lane);
    bf16x16 a0 = load_a_bf16(A, r0,      kb, 256, lane);
    bf16x16 a1 = load_a_bf16(A, r0 + 16, kb, 256, lane);
    acc0 = WMMA_BF16(a0, bf, acc0);
    acc1 = WMMA_BF16(a1, bf, acc1);
  }
  const int n    = col0 + (lane & 15);
  const float gb = gc_bias[n];
  __bf16* out = msg_bf + (size_t)b * 256 * 512;
  const int mb = r0 + (lane >> 4) * 8;
#pragma unroll
  for (int v = 0; v < 8; ++v) {
    out[(size_t)(mb + v) * 512 + n]      = (__bf16)(acc0[v] + gb);
    out[(size_t)(mb + 16 + v) * 512 + n] = (__bf16)(acc1[v] + gb);
  }
}

// ---------------------------------------------------------------------------
// Kernel B: fused GRU. One 16x16 tile of hnew per wave; six accumulators
// (msg@W and h@U at cols c, c+512, c+1024), gate math in registers.
// 2048 row-tiles x 32 col-tiles = 65536 waves -> 8192 blocks.
// ---------------------------------------------------------------------------
__global__ void __launch_bounds__(256)
ggc_gru(const float* __restrict__ ann, const __bf16* __restrict__ ann_bf,
        const __bf16* __restrict__ msg_bf,
        const __bf16* __restrict__ WT, const __bf16* __restrict__ UT,
        const float* __restrict__ gru_b, __bf16* __restrict__ hnew_bf) {
  const int lane = threadIdx.x & 31;
  const int t    = blockIdx.x * 8 + (threadIdx.x >> 5);
  const int row0 = (t >> 5) * 16;   // 0..32752
  const int c0   = (t & 31) * 16;   // 0..496

  f32x8 xz = {}, xr = {}, xh = {}, hz = {}, hr = {}, hh = {};
#pragma unroll 2
  for (int kb = 0; kb < 512; kb += 32) {
    bf16x16 am = load_a_bf16(msg_bf, row0, kb, 512, lane);
    bf16x16 ah = load_a_bf16(ann_bf, row0, kb, 512, lane);
    bf16x16 b0 = load_b_T(WT, c0,        kb, 512, lane);
    bf16x16 b1 = load_b_T(WT, c0 + 512,  kb, 512, lane);
    bf16x16 b2 = load_b_T(WT, c0 + 1024, kb, 512, lane);
    xz = WMMA_BF16(am, b0, xz);
    xr = WMMA_BF16(am, b1, xr);
    xh = WMMA_BF16(am, b2, xh);
    b0 = load_b_T(UT, c0,        kb, 512, lane);
    b1 = load_b_T(UT, c0 + 512,  kb, 512, lane);
    b2 = load_b_T(UT, c0 + 1024, kb, 512, lane);
    hz = WMMA_BF16(ah, b0, hz);
    hr = WMMA_BF16(ah, b1, hr);
    hh = WMMA_BF16(ah, b2, hh);
  }
  const int n = c0 + (lane & 15);
  const float b0z = gru_b[n],        b0r = gru_b[512 + n],        b0h = gru_b[1024 + n];
  const float b1z = gru_b[1536 + n], b1r = gru_b[1536 + 512 + n], b1h = gru_b[1536 + 1024 + n];
  const int mb = row0 + (lane >> 4) * 8;
#pragma unroll
  for (int v = 0; v < 8; ++v) {
    const int m   = mb + v;
    const float h  = ann[(size_t)m * 512 + n];   // f32 state for accurate z*h
    const float z  = sigmoidf_((xz[v] + b0z) + (hz[v] + b1z));
    const float r  = sigmoidf_((xr[v] + b0r) + (hr[v] + b1r));
    const float hc = tanhf((xh[v] + b0h) + r * (hh[v] + b1h));
    hnew_bf[(size_t)m * 512 + n] = (__bf16)(z * h + (1.0f - z) * hc);
  }
}

// ---------------------------------------------------------------------------
// Kernel C: out = hnew(bf16) @ dense_W (via dW^T) + dense_b.  f32 output.
// 32x16 tile per wave. 1024 row groups x 16 col tiles = 16384 waves.
// ---------------------------------------------------------------------------
__global__ void __launch_bounds__(256)
ggc_dense(const __bf16* __restrict__ hnew_bf, const __bf16* __restrict__ dWT,
          const float* __restrict__ dense_b, float* __restrict__ out) {
  const int lane = threadIdx.x & 31;
  const int t    = blockIdx.x * 8 + (threadIdx.x >> 5);
  const int r0   = (t >> 4) * 32;
  const int col0 = (t & 15) * 16;

  f32x8 acc0 = {}, acc1 = {};
#pragma unroll 4
  for (int kb = 0; kb < 512; kb += 32) {
    bf16x16 bf = load_b_T(dWT, col0, kb, 512, lane);
    bf16x16 a0 = load_a_bf16(hnew_bf, r0,      kb, 512, lane);
    bf16x16 a1 = load_a_bf16(hnew_bf, r0 + 16, kb, 512, lane);
    acc0 = WMMA_BF16(a0, bf, acc0);
    acc1 = WMMA_BF16(a1, bf, acc1);
  }
  const int n    = col0 + (lane & 15);
  const float db = dense_b[n];
  const int mb = r0 + (lane >> 4) * 8;
#pragma unroll
  for (int v = 0; v < 8; ++v) {
    out[(size_t)(mb + v) * 256 + n]      = acc0[v] + db;
    out[(size_t)(mb + 16 + v) * 256 + n] = acc1[v] + db;
  }
}

// ---------------------------------------------------------------------------
extern "C" void kernel_launch(void* const* d_in, const int* in_sizes, int n_in,
                              void* d_out, int out_size, void* d_ws, size_t ws_size,
                              hipStream_t stream) {
  const float* adjacent = (const float*)d_in[0];  // [128,256,256]
  const float* ann      = (const float*)d_in[1];  // [128,256,512]
  const float* gc_bias  = (const float*)d_in[2];  // [512]
  const float* gru_W    = (const float*)d_in[3];  // [512,1536]
  const float* gru_U    = (const float*)d_in[4];  // [512,1536]
  const float* gru_b    = (const float*)d_in[5];  // [2,1536]
  const float* dense_W  = (const float*)d_in[6];  // [512,256]
  const float* dense_b  = (const float*)d_in[7];  // [256]
  float* out = (float*)d_out;                     // [128,256,256]

  // Workspace layout (bf16 tensors), ~154 MB total (fits the 192MB L2):
  char* ws = (char*)d_ws;
  const size_t N_ADJ  = (size_t)128 * 256 * 256;   // 8.4M elems
  const size_t N_ANN  = (size_t)128 * 256 * 512;   // 16.8M elems
  const size_t SZ_ADJ = N_ADJ * 2;
  const size_t SZ_ANN = N_ANN * 2;
  const size_t SZ_WT  = (size_t)1536 * 512 * 2;
  const size_t SZ_DWT = (size_t)256 * 512 * 2;

  __bf16* adj_bf  = (__bf16*)(ws);
  __bf16* ann_bf  = (__bf16*)(ws + SZ_ADJ);
  __bf16* hT      = (__bf16*)(ws + SZ_ADJ + SZ_ANN);
  __bf16* WT      = (__bf16*)(ws + SZ_ADJ + 2 * SZ_ANN);
  __bf16* UT      = (__bf16*)(ws + SZ_ADJ + 2 * SZ_ANN + SZ_WT);
  __bf16* dWT     = (__bf16*)(ws + SZ_ADJ + 2 * SZ_ANN + 2 * SZ_WT);
  __bf16* msg_bf  = (__bf16*)(ws + SZ_ADJ + 2 * SZ_ANN + 2 * SZ_WT + SZ_DWT);
  __bf16* hnew_bf = (__bf16*)(ws + SZ_ADJ + 3 * SZ_ANN + 2 * SZ_WT + SZ_DWT);

  // ---- stage 0: precision staging ----------------------------------------
  ggc_cvt_bf16<<<(int)(N_ADJ / 8 / 256), 256, 0, stream>>>(adjacent, adj_bf, N_ADJ);
  ggc_cvt_bf16<<<(int)(N_ANN / 8 / 256), 256, 0, stream>>>(ann, ann_bf, N_ANN);

  dim3 tb(32, 8, 1);
  ggc_transpose_bf16<<<dim3(512 / 32, 256 / 32, 128), tb, 0, stream>>>(ann, hT, 256, 512);
  ggc_transpose_bf16<<<dim3(1536 / 32, 512 / 32, 1), tb, 0, stream>>>(gru_W, WT, 512, 1536);
  ggc_transpose_bf16<<<dim3(1536 / 32, 512 / 32, 1), tb, 0, stream>>>(gru_U, UT, 512, 1536);
  ggc_transpose_bf16<<<dim3(256 / 32, 512 / 32, 1), tb, 0, stream>>>(dense_W, dWT, 512, 256);

  // ---- stage 1..3: WMMA GEMM pipeline (8 wave32 per block) ---------------
  ggc_gconv<<<32768 / 8, 256, 0, stream>>>(adj_bf, hT, gc_bias, msg_bf);
  ggc_gru<<<65536 / 8, 256, 0, stream>>>(ann, ann_bf, msg_bf, WT, UT, gru_b, hnew_bf);
  ggc_dense<<<16384 / 8, 256, 0, stream>>>(hnew_bf, dWT, dense_b, out);
}
